// DeepGate3_26611617366156
// MI455X (gfx1250) — matrix operand order repaired and checked by reading, not compiled, not run on hardware
//
#include <hip/hip_runtime.h>
#include <hip/hip_bf16.h>

// ---------------------------------------------------------------------------
// DeepGate3 forward on MI455X (gfx1250).
// All large GEMMs use v_wmma_f32_16x16x32_f16 (f32 -> f16 operands, f32 acc).
// ---------------------------------------------------------------------------

typedef _Float16 v16h __attribute__((ext_vector_type(16)));
typedef float    v8f  __attribute__((ext_vector_type(8)));

#define D_MODEL 128
#define N_NODES 131072
#define N_HOPS  8192
#define N_PAIRS 262144
#define TOK     (N_HOPS * 10)   // 81920 tokens in the pooled sequences

// ---------------------------------------------------------------------------
// Tiled WMMA GEMM:  C[M,N] = A[M,K] @ W^T + bias   (W is (N,K) row-major)
// Block: 256 threads = 8 waves. C tile 128x64: wave w owns the 16-row strip
// m = w*16 and all four 16-col subtiles -> 4 WMMAs per K-step, A-frag reused.
// Both LDS tiles are stored K-contiguous per row ([row][k] / [n][k]) so every
// fragment load is 2x ds_load_b128. All B fragments are loaded before the
// WMMAs are issued so the DS waits can be partial and overlap with compute.
// Requires: M%128==0, N%64==0, K%32==0 (true for every call site here).
// ---------------------------------------------------------------------------
__global__ __launch_bounds__(256) void gemm_wmma_bt(
    const float* __restrict__ A, int lda,
    const float* __restrict__ W,          // (N,K) row-major
    const float* __restrict__ bias,       // (N)
    float* __restrict__ C, int ldc,
    int M, int N, int K, int relu)
{
    const int bm   = blockIdx.x * 128;
    const int bn   = blockIdx.y * 64;
    const int tid  = threadIdx.x;
    const int wave = tid >> 5;
    const int lane = tid & 31;

    __shared__ _Float16 As[128][40];   // [m][k] 128x32 tile (+pad)
    __shared__ _Float16 Bs[64][40];    // [n][k] 64x32 tile  (+pad)

    v8f acc[4] = {};

    for (int k0 = 0; k0 < K; k0 += 32) {
        __syncthreads();
        // ---- load A tile (128x32), f32 -> f16, K-contiguous ----
        {
            const int r = tid >> 1;
            const int c = (tid & 1) * 16;
            const float* src = A + (size_t)(bm + r) * (size_t)lda + k0 + c;
#pragma unroll
            for (int i = 0; i < 16; ++i) As[r][c + i] = (_Float16)src[i];
        }
        // ---- load B tile transposed: Bs[n][k] = W[(bn+n)*K + k0+k] ----
        {
            const int n = tid >> 2;
            const int c = (tid & 3) * 8;
            const float* src = W + (size_t)(bn + n) * (size_t)K + k0 + c;
#pragma unroll
            for (int i = 0; i < 8; ++i) Bs[n][c + i] = (_Float16)src[i];
        }
        // prefetch next K tile into cache while we compute this one
        if (k0 + 32 < K) {
            __builtin_prefetch(A + (size_t)(bm + (tid >> 1)) * (size_t)lda + k0 + 32, 0, 1);
            __builtin_prefetch(W + (size_t)(bn + (tid >> 2)) * (size_t)K + k0 + 32, 0, 1);
        }
        __syncthreads();

        // ---- A fragment (16x32, CDNA5 layout) ----
        v16h a;
        {
            const int row   = wave * 16 + (lane & 15);
            const int khalf = (lane >= 16) ? 8 : 0;
#pragma unroll
            for (int e = 0; e < 16; ++e) {
                const int v = e >> 1, h = e & 1;
                const int k = ((v >= 4) ? 16 : 0) + khalf + ((v & 3) << 1) + h;
                a[e] = As[row][k];
            }
        }
        // ---- load ALL B fragments first (lets waits be partial) ----
        v16h b[4];
        {
            const int col = lane & 15;
            const int kb  = (lane >= 16) ? 8 : 0;
#pragma unroll
            for (int j = 0; j < 4; ++j) {
#pragma unroll
                for (int e = 0; e < 16; ++e) b[j][e] = Bs[j * 16 + col][kb + (e & 7) + ((e >> 3) << 4)];
            }
        }
        // ---- 4 WMMAs sharing the A fragment ----
#pragma unroll
        for (int j = 0; j < 4; ++j) {
            acc[j] = __builtin_amdgcn_wmma_f32_16x16x32_f16(
                false, a, false, b[j], (short)0, acc[j], false, false);
        }
    }

    // ---- epilogue: bias (+ReLU), store f32 ----
    const int nloc = lane & 15;
    const int mhi  = (lane >= 16) ? 8 : 0;
#pragma unroll
    for (int j = 0; j < 4; ++j) {
#pragma unroll
        for (int r = 0; r < 8; ++r) {
            const int row = bm + wave * 16 + r + mhi;
            const int col = bn + j * 16 + nloc;
            float v = acc[j][r] + bias[col];
            if (relu) v = fmaxf(v, 0.f);
            C[(size_t)row * (size_t)ldc + col] = v;
        }
    }
}

// ---------------------------------------------------------------------------
// In-place LayerNorm over rows (wave per row). Optional ReLU after LN.
// N must be a multiple of 32 (256 or 512 here).
// ---------------------------------------------------------------------------
__global__ __launch_bounds__(256) void ln_rows_kernel(
    float* __restrict__ X, const float* __restrict__ g, const float* __restrict__ be,
    int M, int N, int relu_after)
{
    const int row  = blockIdx.x * 8 + (threadIdx.x >> 5);
    const int lane = threadIdx.x & 31;
    if (row >= M) return;
    float* x = X + (size_t)row * (size_t)N;
    const int per = N >> 5;
    float s = 0.f, s2 = 0.f;
    for (int i = 0; i < per; ++i) {
        const float v = x[lane + i * 32];
        s += v; s2 += v * v;
    }
    for (int off = 16; off > 0; off >>= 1) {
        s  += __shfl_xor(s,  off, 32);
        s2 += __shfl_xor(s2, off, 32);
    }
    const float m   = s / (float)N;
    const float var = s2 / (float)N - m * m;
    const float inv = rsqrtf(var + 1e-5f);
    for (int i = 0; i < per; ++i) {
        const int c = lane + i * 32;
        float v = (x[c] - m) * inv * g[c] + be[c];
        if (relu_after) v = fmaxf(v, 0.f);
        x[c] = v;
    }
}

// ---------------------------------------------------------------------------
// X = LN(X + O) with N=128 (wave per row) — encoder residual LayerNorm.
// ---------------------------------------------------------------------------
__global__ __launch_bounds__(256) void ln_residual_kernel(
    float* __restrict__ X, const float* __restrict__ O,
    const float* __restrict__ g, const float* __restrict__ be, int M)
{
    const int row  = blockIdx.x * 8 + (threadIdx.x >> 5);
    const int lane = threadIdx.x & 31;
    if (row >= M) return;
    float* x = X + (size_t)row * 128;
    const float* o = O + (size_t)row * 128;
    float v[4];
    float s = 0.f, s2 = 0.f;
#pragma unroll
    for (int i = 0; i < 4; ++i) {
        v[i] = x[lane + i * 32] + o[lane + i * 32];
        s += v[i]; s2 += v[i] * v[i];
    }
    for (int off = 16; off > 0; off >>= 1) {
        s  += __shfl_xor(s,  off, 32);
        s2 += __shfl_xor(s2, off, 32);
    }
    const float m   = s * (1.f / 128.f);
    const float var = s2 * (1.f / 128.f) - m * m;
    const float inv = rsqrtf(var + 1e-5f);
#pragma unroll
    for (int i = 0; i < 4; ++i) {
        const int c = lane + i * 32;
        x[c] = (v[i] - m) * inv * g[c] + be[c];
    }
}

// ---------------------------------------------------------------------------
// Final tiny linear (NOUT = 1 or 3): thread per row.
// ---------------------------------------------------------------------------
__global__ void mlp_out_kernel(
    const float* __restrict__ X, const float* __restrict__ W,
    const float* __restrict__ b, float* __restrict__ out,
    int M, int K, int NOUT)
{
    const int row = blockIdx.x * blockDim.x + threadIdx.x;
    if (row >= M) return;
    const float* x = X + (size_t)row * (size_t)K;
    for (int o = 0; o < NOUT; ++o) {
        const float* w = W + (size_t)o * (size_t)K;
        float acc = 0.f;
        for (int k = 0; k < K; ++k) acc += x[k] * w[k];
        out[(size_t)row * NOUT + o] = acc + b[o];
    }
}

// ---------------------------------------------------------------------------
// Attention: S=10, 4 heads x 32 dims. Block = 128 thr (wave per head).
// QKV layout per token: [q(128) k(128) v(128)]. pad_add ADDED to logits
// (replicating the reference exactly: s = s + (1 - mask)).
// ---------------------------------------------------------------------------
__global__ __launch_bounds__(128) void attention_kernel(
    const float* __restrict__ QKV, const float* __restrict__ pad,
    float* __restrict__ Out)
{
    const int hop = blockIdx.x;
    __shared__ float sq[10 * 384];
    for (int i = threadIdx.x; i < 3840; i += 128)
        sq[i] = QKV[(size_t)hop * 3840 + i];
    __syncthreads();

    const int head = threadIdx.x >> 5;
    const int lane = threadIdx.x & 31;
    if (lane < 10) {
        const int i = lane;
        const float* q = &sq[i * 384 + head * 32];
        float sc[10];
        float mx = -1e30f;
#pragma unroll
        for (int j = 0; j < 10; ++j) {
            const float* kk = &sq[j * 384 + 128 + head * 32];
            float d = 0.f;
#pragma unroll
            for (int t = 0; t < 32; ++t) d += q[t] * kk[t];
            d = d * 0.17677669529663687f + pad[hop * 10 + j];
            sc[j] = d;
            mx = fmaxf(mx, d);
        }
        float sum = 0.f;
#pragma unroll
        for (int j = 0; j < 10; ++j) { sc[j] = __expf(sc[j] - mx); sum += sc[j]; }
        const float invs = 1.f / sum;
        float o[32];
#pragma unroll
        for (int t = 0; t < 32; ++t) o[t] = 0.f;
#pragma unroll
        for (int j = 0; j < 10; ++j) {
            const float a = sc[j] * invs;
            const float* vv = &sq[j * 384 + 256 + head * 32];
#pragma unroll
            for (int t = 0; t < 32; ++t) o[t] += a * vv[t];
        }
        float* op = Out + ((size_t)hop * 10 + i) * 128 + head * 32;
#pragma unroll
        for (int t = 0; t < 32; ++t) op[t] = o[t];
    }
}

// ---------------------------------------------------------------------------
// _build_hops: seq (N_HOPS,10,128) + pad_add (N_HOPS,10). Block per hop,
// thread per feature dim (128 threads).
// ---------------------------------------------------------------------------
__global__ __launch_bounds__(128) void build_hops_kernel(
    const float* __restrict__ emb,
    const int* __restrict__ hop_pi, const int* __restrict__ stats,
    const int* __restrict__ hop_po,
    const float* __restrict__ cls, const float* __restrict__ dc,
    const float* __restrict__ zero, const float* __restrict__ one,
    const float* __restrict__ pos,
    float* __restrict__ seq, float* __restrict__ pad, int special)
{
    const int h = blockIdx.x;
    const int d = threadIdx.x;
    int st[8], pi[8];
#pragma unroll
    for (int j = 0; j < 8; ++j) {
        st[j] = stats[h * 8 + j];
        pi[j] = hop_pi[h * 8 + j];
    }
    int kvalid = 0;
#pragma unroll
    for (int j = 0; j < 8; ++j) kvalid += (st[j] != -1);
    const int dc_n = special ? max(0, 6 - kvalid) : 0;

    float slot[8];
#pragma unroll
    for (int p = 0; p < 8; ++p) slot[p] = (p < dc_n) ? dc[d] : 0.f;

    int r = 0;
#pragma unroll
    for (int j = 0; j < 8; ++j) {
        if (st[j] != -1) {
            const int p = dc_n + r; ++r;
            float tok;
            if (special && st[j] == 0)      tok = zero[d];
            else if (special && st[j] == 1) tok = one[d];
            else                            tok = emb[(size_t)pi[j] * 128 + d];
            slot[p] += tok;
        }
    }

    float* srow = seq + (size_t)h * 10 * 128;
    srow[d] = cls[d] + pos[d];                              // token 0 = cls
#pragma unroll
    for (int p = 0; p < 8; ++p)
        srow[(1 + p) * 128 + d] = slot[p] + pos[(1 + p) * 128 + d];
    srow[9 * 128 + d] = emb[(size_t)hop_po[h] * 128 + d] + pos[9 * 128 + d];

    if (d < 10) {
        float mask;
        if (d == 0 || d == 9) mask = 1.f;
        else                  mask = ((d - 1) < (dc_n + kvalid)) ? 1.f : 0.f;
        pad[h * 10 + d] = 1.f - mask;   // reference adds (1-mask) to logits
    }
}

// ---------------------------------------------------------------------------
// connect-pair gather with the reference's interleaved layout:
// row p: [hs[i0,0], hs[i1,0], hs[i0,1], hs[i1,1], ...]
// ---------------------------------------------------------------------------
__global__ __launch_bounds__(128) void gather_gates_kernel(
    const float* __restrict__ hs, const int* __restrict__ cpi,
    float* __restrict__ gates)
{
    const int p = blockIdx.x;
    const int d = threadIdx.x;
    const int i0 = cpi[p];
    const int i1 = cpi[N_PAIRS + p];
    float* g = gates + (size_t)p * 256;
    g[2 * d]     = hs[(size_t)i0 * 128 + d];
    g[2 * d + 1] = hs[(size_t)i1 * 128 + d];
}

// ===========================================================================
// Host orchestration.
//
// Input flattening assumption: top-level dict in insertion order
// (hf, hs, hop_pi, hop_pi_stats, hop_po, connect_pair_index, params), with
// `params` flattened as a JAX pytree (dict keys sorted, lists/tuples in
// order). Resulting indices:
//   6 cls | 7..16 connect_head (W,b,g,be)x2,(W,b) | 17 dc
//   18..23 hop_head (W1,W2,b1,b2,be,g) | 24 one | 25 pos_emb
//   26..35 readout_level | 36..45 readout_num | 46..55 readout_prob
//   56..91 tf x3 (W1,W2,Wo,Wqkv,b1,b2,be1,be2,bo,bqkv,g1,g2) | 92 zero
// ===========================================================================
extern "C" void kernel_launch(void* const* d_in, const int* in_sizes, int n_in,
                              void* d_out, int out_size, void* d_ws, size_t ws_size,
                              hipStream_t stream)
{
    (void)in_sizes; (void)n_in; (void)out_size; (void)ws_size;

    const float* hf = (const float*)d_in[0];
    const float* hs = (const float*)d_in[1];
    const int* hop_pi = (const int*)d_in[2];
    const int* hop_st = (const int*)d_in[3];
    const int* hop_po = (const int*)d_in[4];
    const int* cpi    = (const int*)d_in[5];
    auto P = [&](int i) { return (const float*)d_in[i]; };

    const int P_CLS = 6, P_CH = 7, P_DC = 17;
    const int P_HH_W1 = 18, P_HH_W2 = 19, P_HH_B1 = 20, P_HH_B2 = 21,
              P_HH_BE = 22, P_HH_G = 23;
    const int P_ONE = 24, P_POS = 25;
    const int P_RL = 26, P_RN = 36, P_RP = 46, P_TF = 56, P_ZERO = 92;

    // ---- output layout (return order, flat) ----
    float* out = (float*)d_out;
    float* o_hs      = out;
    float* o_hf      = o_hs + (size_t)N_NODES * 128;
    float* o_prob    = o_hf + (size_t)N_NODES * 128;
    float* o_hoptt   = o_prob + N_NODES;
    float* o_level   = o_hoptt + (size_t)N_HOPS * 64;
    float* o_connect = o_level + N_NODES;
    float* o_hopnum  = o_connect + (size_t)N_PAIRS * 3;

    // ---- workspace layout ----
    char* w = (char*)d_ws;
    float* wsA = (float*)w;                                   // 671 MB
    w += (size_t)TOK * 2048 * sizeof(float);
    float* wsB = (float*)w;                                   // 268 MB
    w += (size_t)N_PAIRS * 256 * sizeof(float);
    float* seq = (float*)w;                                   // 40 MB
    w += (size_t)N_HOPS * 10 * 128 * sizeof(float);
    float* pad = (float*)w;                                   // 0.3 MB
    float* oproj = wsB + (size_t)TOK * 128;                   // 2nd chunk of wsB

    // ---- launch helpers ----
    auto gemm = [&](const float* A, int lda, const float* W, const float* bias,
                    float* C, int ldc, int M, int N, int K, int relu) {
        dim3 grid(M / 128, N / 64);
        gemm_wmma_bt<<<grid, 256, 0, stream>>>(A, lda, W, bias, C, ldc, M, N, K, relu);
    };
    auto ln = [&](float* X, const float* g_, const float* b_, int M, int N, int relu) {
        ln_rows_kernel<<<(M + 7) / 8, 256, 0, stream>>>(X, g_, b_, M, N, relu);
    };
    auto lnres = [&](float* X, const float* O, const float* g_, const float* b_, int M) {
        ln_residual_kernel<<<(M + 7) / 8, 256, 0, stream>>>(X, O, g_, b_, M);
    };
    // 3-layer readout MLP: K0 -> 256 -> 256 -> NOUT (LN+ReLU between layers)
    auto run_mlp = [&](const float* X, int lda, int M, int K0, int pb,
                       float* outp, int nout) {
        gemm(X, lda, P(pb + 0), P(pb + 1), wsA, 256, M, 256, K0, 0);
        ln(wsA, P(pb + 2), P(pb + 3), M, 256, 1);
        gemm(wsA, 256, P(pb + 4), P(pb + 5), wsB, 256, M, 256, 256, 0);
        ln(wsB, P(pb + 6), P(pb + 7), M, 256, 1);
        mlp_out_kernel<<<(M + 255) / 256, 256, 0, stream>>>(
            wsB, P(pb + 8), P(pb + 9), outp, M, 256, nout);
    };
    // 3-layer transformer encoder over seq (TOK x 128)
    auto run_pool = [&]() {
        for (int l = 0; l < 3; ++l) {
            const int t = P_TF + l * 12;
            // QKV projection
            gemm(seq, 128, P(t + 3), P(t + 9), wsA, 384, TOK, 384, 128, 0);
            // attention -> wsB (TOK x 128)
            attention_kernel<<<N_HOPS, 128, 0, stream>>>(wsA, pad, wsB);
            // output projection
            gemm(wsB, 128, P(t + 2), P(t + 8), oproj, 128, TOK, 128, 128, 0);
            lnres(seq, oproj, P(t + 10), P(t + 6), TOK);        // LN(x + o)
            // FFN
            gemm(seq, 128, P(t + 0), P(t + 4), wsA, 2048, TOK, 2048, 128, 1);
            gemm(wsA, 2048, P(t + 1), P(t + 5), wsB, 128, TOK, 128, 2048, 0);
            lnres(seq, wsB, P(t + 11), P(t + 7), TOK);          // LN(x + f)
        }
    };

    // ---- P0: passthrough outputs ----
    hipMemcpyAsync(o_hs, hs, (size_t)N_NODES * 128 * sizeof(float),
                   hipMemcpyDeviceToDevice, stream);
    hipMemcpyAsync(o_hf, hf, (size_t)N_NODES * 128 * sizeof(float),
                   hipMemcpyDeviceToDevice, stream);

    // ---- P1/P2: node readouts ----
    run_mlp(hf, 128, N_NODES, 128, P_RP, o_prob, 1);
    run_mlp(hs, 128, N_NODES, 128, P_RL, o_level, 1);

    // ---- P3: connect head ----
    gather_gates_kernel<<<N_PAIRS, 128, 0, stream>>>(hs, cpi, wsB);
    // layer0 reads wsB (gates), writes wsA; layer1 overwrites wsB — safe.
    run_mlp(wsB, 256, N_PAIRS, 256, P_CH, o_connect, 3);

    // ---- P4: pool over hf + hop head ----
    build_hops_kernel<<<N_HOPS, 128, 0, stream>>>(
        hf, hop_pi, hop_st, hop_po, P(P_CLS), P(P_DC), P(P_ZERO), P(P_ONE),
        P(P_POS), seq, pad, 1);
    run_pool();
    // hop_head on seq[:,0,:] via lda=1280
    gemm(seq, 1280, P(P_HH_W1), P(P_HH_B1), wsB, 512, N_HOPS, 512, 128, 1);
    ln(wsB, P(P_HH_G), P(P_HH_BE), N_HOPS, 512, 0);
    gemm(wsB, 512, P(P_HH_W2), P(P_HH_B2), o_hoptt, 64, N_HOPS, 64, 512, 0);

    // ---- P5: pool over hs + readout_num ----
    build_hops_kernel<<<N_HOPS, 128, 0, stream>>>(
        hs, hop_pi, hop_st, hop_po, P(P_CLS), P(P_DC), P(P_ZERO), P(P_ONE),
        P(P_POS), seq, pad, 0);
    run_pool();
    run_mlp(seq, 1280, N_HOPS, 128, P_RN, o_hopnum, 1);
}
